// CausalAttention_62191126446396
// MI455X (gfx1250) — matrix operand order repaired and checked by reading, not compiled, not run on hardware
//
#include <hip/hip_runtime.h>

// ---------------------------------------------------------------------------
// CDNA5 (gfx1250) causal attention: bf16 WMMA everywhere, fp32 accumulate.
// B=2, S=2048, D=1024, H=16, hd=64.
// ---------------------------------------------------------------------------

typedef __bf16 v8bf  __attribute__((ext_vector_type(8)));
typedef __bf16 v16bf __attribute__((ext_vector_type(16)));
typedef float  v8f   __attribute__((ext_vector_type(8)));
typedef int    v4i   __attribute__((ext_vector_type(4)));

#define BB 2
#define SS 2048
#define DD 1024
#define HH 16
#define HD 64
#define MTOT (BB * SS)          // 4096
#define QKVN (3 * DD)           // 3072

// Async global->LDS (ASYNCcnt) path, guarded: falls back to sync staging.
// Probe-derived signature: (v4i AS1* src, v4i AS3* lds, imm offset, imm cpol).
#if __has_builtin(__builtin_amdgcn_global_load_async_to_lds_b128) && \
    __has_builtin(__builtin_amdgcn_s_wait_asynccnt)
#define USE_ASYNC_LDS 1
#define AS1 __attribute__((address_space(1)))
#define AS3 __attribute__((address_space(3)))
#else
#define USE_ASYNC_LDS 0
#endif

__device__ __forceinline__ v8f wmma_bf16(v16bf a, v16bf b, v8f c) {
  // D = A(16x32) * B(32x16) + C, fp32 accumulate
  return __builtin_amdgcn_wmma_f32_16x16x32_bf16(
      /*neg_a=*/false, a, /*neg_b=*/false, b,
      /*c_mod=*/(short)0, c, /*reuse_a=*/false, /*reuse_b=*/false);
}

// A-matrix (16x32 bf16) fragment: lane -> row M=lane&15, K = 2*hi8 .. 2*hi8+15
__device__ __forceinline__ v16bf load_a_halves(const __bf16* base, int hi8) {
  v8bf lo = *(const v8bf*)(base + 2 * hi8);
  v8bf hi = *(const v8bf*)(base + 2 * hi8 + 8);
  v16bf r;
#pragma unroll
  for (int i = 0; i < 8; ++i) { r[i] = lo[i]; r[i + 8] = hi[i]; }
  return r;
}

// B-matrix (32x16 bf16) fragment: lane -> col N=lane&15, halfs j -> K = j + (lane>=16?16:0).
__device__ __forceinline__ v16bf load_b_contig(const __bf16* p) {
  v8bf lo = *(const v8bf*)(p);
  v8bf hi = *(const v8bf*)(p + 8);
  v16bf r;
#pragma unroll
  for (int i = 0; i < 8; ++i) { r[i] = lo[i]; r[i + 8] = hi[i]; }
  return r;
}

// ---- 16-lane reductions via DPP16 row rotations (VALU, no LDS traffic) ----
template <int N>
__device__ __forceinline__ float row_ror(float x) {
#if __has_builtin(__builtin_amdgcn_update_dpp)
  int i = __builtin_amdgcn_update_dpp(0, __float_as_int(x), 0x120 + N, 0xf, 0xf, true);
  return __int_as_float(i);
#else
  return __shfl_xor(x, N, 32);
#endif
}
__device__ __forceinline__ float rmax16(float v) {
  v = fmaxf(v, row_ror<8>(v));
  v = fmaxf(v, row_ror<4>(v));
  v = fmaxf(v, row_ror<2>(v));
  v = fmaxf(v, row_ror<1>(v));
  return v;
}
__device__ __forceinline__ float rsum16(float v) {
  v += row_ror<8>(v);
  v += row_ror<4>(v);
  v += row_ror<2>(v);
  v += row_ror<1>(v);
  return v;
}

// ---------------------------------------------------------------------------
// fp32 -> bf16 conversions
// ---------------------------------------------------------------------------
__global__ void k_f32_to_bf16(const float* __restrict__ in, __bf16* __restrict__ out, int n) {
  int i = blockIdx.x * blockDim.x + threadIdx.x;
  if (i < n) out[i] = (__bf16)in[i];
}

// in [K][N] fp32 (row-major) -> out [N][K] bf16 (transposed)
__global__ void k_transpose_bf16(const float* __restrict__ in, __bf16* __restrict__ out,
                                 int K, int N) {
  int i = blockIdx.x * blockDim.x + threadIdx.x;
  if (i < K * N) {
    int k = i / N, n = i - k * N;
    out[(size_t)n * K + k] = (__bf16)in[i];
  }
}

// ---------------------------------------------------------------------------
// GEMM: C[M,N] = A[M,K](bf16) @ B[K,N] + bias, with B given transposed BT[N][K].
// Block tile 64(M) x 256(N), BK=32; 8 waves (2x4), each computing 32x64
// (8 WMMAs per K-step per wave).  Double-buffered async global->LDS staging
// when the toolchain exposes the gfx1250 async builtins.
// ---------------------------------------------------------------------------
template <bool OUT_BF16, bool WRITE_VT>
__global__ __launch_bounds__(256) void k_gemm_bf16(
    const __bf16* __restrict__ A, const __bf16* __restrict__ BT,
    const float* __restrict__ bias, float* __restrict__ Cf,
    __bf16* __restrict__ Cb, __bf16* __restrict__ vT, int M, int N, int K) {
#if USE_ASYNC_LDS
  __shared__ __bf16 As[2][64 * 40];
  __shared__ __bf16 Bs[2][256 * 40];
#else
  __shared__ __bf16 As[1][64 * 40];
  __shared__ __bf16 Bs[1][256 * 40];
#endif

  const int tid  = threadIdx.x;
  const int lane = tid & 31;
  const int wave = tid >> 5;         // 0..7
  const int wm   = wave >> 2;        // 0..1  -> 32 rows
  const int wn   = wave & 3;         // 0..3  -> 64 cols
  const int m0   = blockIdx.y * 64;
  const int n0   = blockIdx.x * 256;
  const int l15  = lane & 15;
  const int hi8  = (lane >= 16) ? 8 : 0;
  const int kb16 = (lane >= 16) ? 16 : 0;

  v8f acc[2][4] = {};
  const int NT = K >> 5;             // number of 32-wide K tiles

#if USE_ASYNC_LDS
  // 5 async b128 loads per thread per tile (1 for A, 4 for B), in-order per wave.
  auto issue_tile = [&](int kt) {
    int k0 = kt * 32, buf = kt & 1;
    {
      int row = tid >> 2, cb = (tid & 3) * 8;
      __builtin_amdgcn_global_load_async_to_lds_b128(
          (AS1 v4i*)(A + (size_t)(m0 + row) * K + k0 + cb),
          (AS3 v4i*)(&As[buf][row * 40 + cb]), 0, 0);
    }
#pragma unroll
    for (int i = 0; i < 4; ++i) {
      int li = tid + 256 * i;
      int row = li >> 2, cb = (li & 3) * 8;
      __builtin_amdgcn_global_load_async_to_lds_b128(
          (AS1 v4i*)(BT + (size_t)(n0 + row) * K + k0 + cb),
          (AS3 v4i*)(&Bs[buf][row * 40 + cb]), 0, 0);
    }
  };
  issue_tile(0);
#endif

  for (int kt = 0; kt < NT; ++kt) {
#if USE_ASYNC_LDS
    const int buf = kt & 1;
    if (kt + 1 < NT) {
      issue_tile(kt + 1);                       // overlap next tile with compute
      __builtin_amdgcn_s_wait_asynccnt(5);      // retire previous tile's 5 loads
    } else {
      __builtin_amdgcn_s_wait_asynccnt(0);
    }
    __syncthreads();                            // tile kt visible to all waves
#else
    const int buf = 0;
    const int k0 = kt * 32;
    {
      int row = tid >> 2, cb = (tid & 3) * 8;
      v8bf d = *(const v8bf*)(A + (size_t)(m0 + row) * K + k0 + cb);
      *(v8bf*)(&As[0][row * 40 + cb]) = d;
      if (k0 + 32 < K)
        __builtin_prefetch(A + (size_t)(m0 + row) * K + k0 + 32 + cb, 0, 3);
    }
#pragma unroll
    for (int i = 0; i < 4; ++i) {
      int li = tid + 256 * i;
      int row = li >> 2, cb = (li & 3) * 8;
      v8bf d = *(const v8bf*)(BT + (size_t)(n0 + row) * K + k0 + cb);
      *(v8bf*)(&Bs[0][row * 40 + cb]) = d;
      if (k0 + 32 < K)
        __builtin_prefetch(BT + (size_t)(n0 + row) * K + k0 + 32 + cb, 0, 3);
    }
    __syncthreads();
#endif

    v16bf af[2], bf4[4];
#pragma unroll
    for (int mi = 0; mi < 2; ++mi)
      af[mi] = load_a_halves(&As[buf][(wm * 32 + mi * 16 + l15) * 40], hi8);
#pragma unroll
    for (int ni = 0; ni < 4; ++ni)
      bf4[ni] = load_b_contig(&Bs[buf][(wn * 64 + ni * 16 + l15) * 40 + kb16]);

#pragma unroll
    for (int mi = 0; mi < 2; ++mi)
#pragma unroll
      for (int ni = 0; ni < 4; ++ni)
        acc[mi][ni] = wmma_bf16(af[mi], bf4[ni], acc[mi][ni]);

    __syncthreads();                            // done reading tile kt
  }

  // epilogue: C layout = lane gives N=lane&15, VGPR r gives M = r + hi8
#pragma unroll
  for (int mi = 0; mi < 2; ++mi) {
#pragma unroll
    for (int ni = 0; ni < 4; ++ni) {
      int ncol = n0 + wn * 64 + ni * 16 + l15;
      float bv = bias ? bias[ncol] : 0.0f;
#pragma unroll
      for (int r = 0; r < 8; ++r) {
        int m = m0 + wm * 32 + mi * 16 + r + hi8;
        float val = acc[mi][ni][r] + bv;
        if (OUT_BF16) Cb[(size_t)m * N + ncol] = (__bf16)val;
        else          Cf[(size_t)m * N + ncol] = val;
        if (WRITE_VT) {
          if (ncol >= 2 * DD) {                  // V portion of qkv
            int h = (ncol - 2 * DD) >> 6;
            int c = (ncol - 2 * DD) & 63;
            int b = m >> 11, s = m & (SS - 1);
            vT[(((size_t)(b * HH + h) << 6) + c) * SS + s] = (__bf16)val;
          }
        }
      }
    }
  }
}

// ---------------------------------------------------------------------------
// Flash attention: one wave per (b,h, 32-row q tile) = two 16-row subtiles
// sharing the same K/V fragments (16 WMMAs per 32-key chunk).
// Steady-state chunks are unmasked; exactly one diagonal chunk is masked.
// Q is pre-scaled by 1/8 (exact in bf16).
// ---------------------------------------------------------------------------
__global__ __launch_bounds__(256) void k_attn(const __bf16* __restrict__ qkv,
                                              const __bf16* __restrict__ vT,
                                              __bf16* __restrict__ out) {
  __shared__ __bf16 Pst[8][2][16 * 40];   // per-wave, per-subtile P staging

  const int lane = threadIdx.x & 31;
  const int wave = threadIdx.x >> 5;
  const int wid  = blockIdx.x * 8 + wave;  // 0..2047
  const int qt   = wid & 63;               // 64 x 32-row tiles per (b,h)
  const int bh   = wid >> 6;               // b*16 + h
  const int h    = bh & 15, b = bh >> 4;
  const int q0   = qt * 32;
  const int l15  = lane & 15;
  const int hi8  = (lane >= 16) ? 8 : 0;
  const int kb16 = (lane >= 16) ? 16 : 0;

  const __bf16* base = qkv + (size_t)b * SS * QKVN;  // [s][3072]

  // Q fragments for both subtiles, pre-scaled by 1/sqrt(hd) = 0.125 (exact)
  v16bf aq[2][2];
#pragma unroll
  for (int t = 0; t < 2; ++t) {
    const __bf16* qrow = base + (size_t)(q0 + 16 * t + l15) * QKVN + HD * h;
#pragma unroll
    for (int c = 0; c < 2; ++c) {
      v16bf f = load_a_halves(qrow + 32 * c, hi8);
#pragma unroll
      for (int i = 0; i < 16; ++i) f[i] = (__bf16)(0.125f * (float)f[i]);
      aq[t][c] = f;
    }
  }

  float m[2][8], l[2][8];
#pragma unroll
  for (int t = 0; t < 2; ++t)
#pragma unroll
    for (int r = 0; r < 8; ++r) { m[t][r] = -__builtin_inff(); l[t][r] = 0.0f; }
  v8f o[2][4] = {};

  auto process = [&](int kbase, bool masked) {
    // ---- shared K fragments (B-matrix for Q K^T) and V fragments ----
    const __bf16* krow0 = base + (size_t)(kbase + l15) * QKVN + DD + HD * h;
    const __bf16* krow1 = krow0 + (size_t)16 * QKVN;
    v16bf bk[2][2];
#pragma unroll
    for (int c = 0; c < 2; ++c) {
      bk[0][c] = load_b_contig(krow0 + 32 * c + kb16);
      bk[1][c] = load_b_contig(krow1 + 32 * c + kb16);
    }
    const __bf16* vb = vT + ((size_t)bh << 6) * SS + kbase + kb16;
    v16bf bv[4];
#pragma unroll
    for (int c = 0; c < 4; ++c)
      bv[c] = load_b_contig(vb + (size_t)(16 * c + l15) * SS);

#pragma unroll
    for (int t = 0; t < 2; ++t) {
      v8f s0 = {}, s1 = {};
#pragma unroll
      for (int c = 0; c < 2; ++c) {
        s0 = wmma_bf16(aq[t][c], bk[0][c], s0);
        s1 = wmma_bf16(aq[t][c], bk[1][c], s1);
      }
      // ---- online softmax (row stats per VGPR index r, per 16-lane half) ----
      float alpha[8];
#pragma unroll
      for (int r = 0; r < 8; ++r) {
        float v0 = s0[r], v1 = s1[r];
        if (masked) {
          int qr = q0 + 16 * t + r + hi8;
          if (kbase + l15 > qr)      v0 = -__builtin_inff();
          if (kbase + 16 + l15 > qr) v1 = -__builtin_inff();
        }
        float mn = fmaxf(m[t][r], rmax16(fmaxf(v0, v1)));
        alpha[r] = __expf(m[t][r] - mn);
        float p0 = __expf(v0 - mn);
        float p1 = __expf(v1 - mn);
        s0[r] = p0; s1[r] = p1;
        l[t][r] = l[t][r] * alpha[r] + rsum16(p0 + p1);
        m[t][r] = mn;
      }
#pragma unroll
      for (int c = 0; c < 4; ++c)
#pragma unroll
        for (int r = 0; r < 8; ++r) o[t][c][r] *= alpha[r];

      // ---- P: C-layout -> LDS -> A-layout fragment ----
      __bf16* Pw = &Pst[wave][t][0];
#pragma unroll
      for (int r = 0; r < 8; ++r) {
        int row = r + hi8;
        Pw[row * 40 + l15]      = (__bf16)s0[r];
        Pw[row * 40 + 16 + l15] = (__bf16)s1[r];
      }
      v16bf pa = load_a_halves(Pw + l15 * 40, hi8);

      // ---- O += P @ V ----
#pragma unroll
      for (int c = 0; c < 4; ++c) o[t][c] = wmma_bf16(pa, bv[c], o[t][c]);
    }
  };

  const int kcf = q0 >> 5;             // fully-unmasked chunks
  for (int kc = 0; kc < kcf; ++kc) process(kc * 32, false);
  process(kcf * 32, true);             // single diagonal (masked) chunk

  // ---- epilogue: normalize and write [B][S][D] bf16 ----
#pragma unroll
  for (int t = 0; t < 2; ++t) {
    __bf16* ob = out + ((size_t)(b * SS + q0 + 16 * t)) * DD + HD * h;
#pragma unroll
    for (int c = 0; c < 4; ++c) {
#pragma unroll
      for (int r = 0; r < 8; ++r) {
        int row = r + hi8;
        ob[(size_t)row * DD + 16 * c + l15] = (__bf16)(o[t][c][r] * (1.0f / l[t][r]));
      }
    }
  }
}

// ---------------------------------------------------------------------------
// launch
// ---------------------------------------------------------------------------
extern "C" void kernel_launch(void* const* d_in, const int* in_sizes, int n_in,
                              void* d_out, int out_size, void* d_ws, size_t ws_size,
                              hipStream_t stream) {
  const float* x      = (const float*)d_in[0];
  const float* w_qkv  = (const float*)d_in[1];
  const float* b_qkv  = (const float*)d_in[2];
  const float* w_proj = (const float*)d_in[3];
  const float* b_proj = (const float*)d_in[4];
  float* out = (float*)d_out;

  // workspace layout (56 MB total)
  char* ws = (char*)d_ws;
  __bf16* xb   = (__bf16*)(ws);                        //  8 MB  x bf16 [4096][1024]
  __bf16* wqkT = (__bf16*)(ws + (size_t)8  * 1048576); //  6 MB  w_qkv^T bf16 [3072][1024]
  __bf16* wprT = (__bf16*)(ws + (size_t)14 * 1048576); //  2 MB  w_proj^T bf16 [1024][1024]
  __bf16* qkv  = (__bf16*)(ws + (size_t)16 * 1048576); // 24 MB  qkv bf16 [4096][3072]
  __bf16* vT   = (__bf16*)(ws + (size_t)40 * 1048576); //  8 MB  V^T bf16 [32*64][2048]
  __bf16* attn = (__bf16*)(ws + (size_t)48 * 1048576); //  8 MB  attn out bf16 [4096][1024]

  // 1) conversions
  {
    int n = MTOT * DD;  // 4,194,304
    k_f32_to_bf16<<<(n + 255) / 256, 256, 0, stream>>>(x, xb, n);
  }
  {
    int n = DD * QKVN;  // 3,145,728
    k_transpose_bf16<<<(n + 255) / 256, 256, 0, stream>>>(w_qkv, wqkT, DD, QKVN);
  }
  {
    int n = DD * DD;    // 1,048,576
    k_transpose_bf16<<<(n + 255) / 256, 256, 0, stream>>>(w_proj, wprT, DD, DD);
  }

  // 2) QKV projection (bf16 out + V^T scatter)
  {
    dim3 grid(QKVN / 256, MTOT / 64);  // (12, 64)
    k_gemm_bf16<true, true><<<grid, 256, 0, stream>>>(
        xb, wqkT, b_qkv, nullptr, qkv, vT, MTOT, QKVN, DD);
  }

  // 3) flash attention (one wave per 32-row q tile; 2048 tiles / 8 waves)
  k_attn<<<256, 256, 0, stream>>>(qkv, vT, attn);

  // 4) output projection (fp32 out)
  {
    dim3 grid(DD / 256, MTOT / 64);    // (4, 64)
    k_gemm_bf16<false, false><<<grid, 256, 0, stream>>>(
        attn, wprT, b_proj, out, nullptr, nullptr, MTOT, DD, DD);
  }
}
